// EGNN_dynamics_AD2_cat_79946521248241
// MI455X (gfx1250) — matrix operand order repaired and checked by reading, not compiled, not run on hardware
//
#include <hip/hip_runtime.h>
#include <hip/hip_bf16.h>
#include <math.h>

#define BB 1024
#define PP 22
#define HH 64
#define LL 5
#define NN (BB*PP)            // 22528 nodes
#define EE (BB*PP*(PP-1))     // 473088 directed edges
#define LDK1 160              // edge MLP K padded 130 -> 160
#define LDKN 128              // node MLP K = 2H
#define CRc 3.0f              // coords_range 15/5
#define LOG2Mc 7.0f

// packed f16 transposed weights, per layer (halves):
//  [0,10240)    w1T  64x160
//  [10240,14336) w2T 64x64
//  [14336,18432) c1T 64x64
//  [18432,26624) nw1T 64x128
//  [26624,30720) nw2T 64x64
#define WPL 30720
#define EDGE_WB 18432         // halves staged by edge kernel (36864 B)
#define NODE_WB 12288         // halves staged by node kernel (24576 B)

typedef __attribute__((ext_vector_type(4)))  _Float16 v4h;
typedef __attribute__((ext_vector_type(8)))  _Float16 v8h;
typedef __attribute__((ext_vector_type(16))) _Float16 v16h;
typedef __attribute__((ext_vector_type(8)))  float    v8f;
typedef __attribute__((ext_vector_type(4)))  unsigned int v4u;
typedef __attribute__((ext_vector_type(8)))  int      v8i;
typedef __attribute__((ext_vector_type(4)))  int      v4i;

__device__ __forceinline__ float siluf(float x) { return x / (1.f + __expf(-x)); }
__device__ __forceinline__ float sigmf(float x) { return 1.f / (1.f + __expf(-x)); }

// ---------------- CDNA5 async / TDM helpers ----------------
template<int OFF>
__device__ __forceinline__ void async_b128(unsigned lds, unsigned long long gaddr) {
  asm volatile("global_load_async_to_lds_b128 %0, %1, off offset:%2"
               :: "v"(lds), "v"(gaddr), "n"(OFF) : "memory");
}
// copy 128 contiguous bytes global -> LDS (one h16 row), per lane
__device__ __forceinline__ void async_row128(unsigned lds, unsigned long long g) {
  async_b128<0>(lds, g);  async_b128<16>(lds, g); async_b128<32>(lds, g);
  async_b128<48>(lds, g); async_b128<64>(lds, g); async_b128<80>(lds, g);
  async_b128<96>(lds, g); async_b128<112>(lds, g);
}
__device__ __forceinline__ void wait_async0() {
  asm volatile("s_wait_asynccnt 0x0" ::: "memory");
}
// TDM: DMA n8*8 contiguous bytes from gsrc into LDS at byte offset ldsOff.
// D# per CDNA5 ISA: 1-row tile, data_size=8B, type=2.
// This toolchain uses the 6-arg builtin: (v4u g0, v8i g1, v4i g2, v4i g3, v8i g4, i32 cpol).
__device__ __forceinline__ void tdm_load(const _Float16* gsrc, unsigned ldsOff, int n8) {
  unsigned long long ga = (unsigned long long)gsrc;
  v4u g0 = { 1u,                                   // count=1 (valid descriptor)
             ldsOff,                               // lds_addr
             (unsigned)ga,                         // global_addr[31:0]
             ((unsigned)(ga >> 32) & 0x01FFFFFFu) | 0x80000000u }; // addr[56:32] | type=2
  v8i g1 = { (int)0x00030000,                      // wg_mask=0, data_size=8B
             (int)(((unsigned)n8 & 0xFFFFu) << 16),// tensor_dim0 lo16 (<<16)
             (int)(((unsigned)n8 >> 16) | (1u << 16)), // tensor_dim0 hi16 | tensor_dim1=1
             (int)(((unsigned)n8 & 0xFFFFu) << 16),// tile_dim0
             1,                                    // tile_dim1=1
             n8,                                   // tensor_dim0_stride lo32
             0, 0 };
  v4i gz4 = { 0, 0, 0, 0 };
  v8i gz8 = { 0, 0, 0, 0, 0, 0, 0, 0 };
  __builtin_amdgcn_tensor_load_to_lds(g0, g1, gz4, gz4, gz8, 0);
}
__device__ __forceinline__ unsigned lds_off(const void* p) {
  return (unsigned)(unsigned long long)p;   // low 32 bits of generic ptr = LDS offset
}

// ---- WMMA fragment loaders (16x16x32 f16, wave32; A and B both use contiguous k = 16*(lane/16) + s) ----
__device__ __forceinline__ v16h loadA(const _Float16* T, int ldk, int lane, int kc) {
  int row = lane & 15;
  int kb  = kc * 32 + ((lane & 16) ? 16 : 0);
  v8h lo = *(const v8h*)(T + row * ldk + kb);
  v8h hi = *(const v8h*)(T + row * ldk + kb + 8);
  return __builtin_shufflevector(lo, hi, 0,1,2,3,4,5,6,7,8,9,10,11,12,13,14,15);
}
__device__ __forceinline__ v16h loadB(const _Float16* W, int ldk, int lane, int nt, int kc) {
  int col = nt * 16 + (lane & 15);
  int kb  = kc * 32 + ((lane & 16) ? 16 : 0);
  return *(const v16h*)(W + col * ldk + kb);
}
__device__ __forceinline__ void zero_acc(v8f acc[4]) {
  v8f z = {0.f,0.f,0.f,0.f,0.f,0.f,0.f,0.f};
  acc[0] = z; acc[1] = z; acc[2] = z; acc[3] = z;
}
template<int LDA, int LDW, int KC>
__device__ __forceinline__ void wave_gemm(const _Float16* A, const _Float16* W,
                                          int lane, v8f acc[4]) {
#pragma unroll
  for (int kc = 0; kc < KC; ++kc) {
    v16h a = loadA(A, LDA, lane, kc);
#pragma unroll
    for (int nt = 0; nt < 4; ++nt) {
      v16h b = loadB(W, LDW, lane, nt, kc);
      acc[nt] = __builtin_amdgcn_wmma_f32_16x16x32_f16(
          false, a, false, b, (short)0, acc[nt], false, false);
    }
  }
}

// ----------------------------- prep / init kernels --------------------------
__global__ void prep_weights_kernel(const float* __restrict__ ew1, const float* __restrict__ ew2,
                                    const float* __restrict__ cw1, const float* __restrict__ nw1,
                                    const float* __restrict__ nw2, _Float16* __restrict__ wpack) {
  int idx = blockIdx.x * blockDim.x + threadIdx.x;
  if (idx >= LL * WPL) return;
  int l = idx / WPL, o = idx - l * WPL;
  _Float16 val;
  if (o < 10240)      { int n = o / 160, k = o - n*160;
                        val = (k < 130) ? (_Float16)ew1[(l*130 + k)*HH + n] : (_Float16)0.f; }
  else if (o < 14336) { int q = o - 10240; int n = q >> 6, k = q & 63;
                        val = (_Float16)ew2[(l*64 + k)*HH + n]; }
  else if (o < 18432) { int q = o - 14336; int n = q >> 6, k = q & 63;
                        val = (_Float16)cw1[(l*64 + k)*HH + n]; }
  else if (o < 26624) { int q = o - 18432; int n = q >> 7, k = q & 127;
                        val = (_Float16)nw1[(l*128 + k)*HH + n]; }
  else                { int q = o - 26624; int n = q >> 6, k = q & 63;
                        val = (_Float16)nw2[(l*64 + k)*HH + n]; }
  wpack[idx] = val;
}

__global__ void init_h_kernel(const float* __restrict__ t, const float* __restrict__ embW,
                              const float* __restrict__ embb,
                              float* __restrict__ h, _Float16* __restrict__ h16) {
  int idx = blockIdx.x * blockDim.x + threadIdx.x;
  if (idx >= NN * HH) return;
  int n = idx >> 6, c = idx & 63;
  int atom = n % PP, b = n / PP;
  float v = embW[atom*HH + c] + t[b]*embW[22*HH + c] + LOG2Mc*embW[23*HH + c] + embb[c];
  h[idx] = v;
  h16[idx] = (_Float16)v;
}

__global__ void init_coord_kernel(const float* __restrict__ x, float* __restrict__ coord) {
  int idx = blockIdx.x * blockDim.x + threadIdx.x;
  if (idx < NN * 3) coord[idx] = x[idx];
}

__global__ void init_eattr_kernel(const float* __restrict__ x, const int* __restrict__ rows,
                                  const int* __restrict__ cols, float* __restrict__ eattr) {
  int e = blockIdx.x * blockDim.x + threadIdx.x;
  if (e >= EE) return;
  int r = rows[e], c = cols[e];
  float dx = x[r*3+0] - x[c*3+0];
  float dy = x[r*3+1] - x[c*3+1];
  float dz = x[r*3+2] - x[c*3+2];
  eattr[e] = dx*dx + dy*dy + dz*dz;
}

__global__ void zero_kernel(float* __restrict__ agg, float* __restrict__ dcoord) {
  int idx = blockIdx.x * blockDim.x + threadIdx.x;
  if (idx < NN * HH)            agg[idx] = 0.f;
  else if (idx < NN*HH + NN*3)  dcoord[idx - NN*HH] = 0.f;
}

__global__ void apply_coord_kernel(float* __restrict__ coord, const float* __restrict__ dcoord) {
  int idx = blockIdx.x * blockDim.x + threadIdx.x;
  if (idx < NN * 3) coord[idx] += dcoord[idx];
}

__global__ void finalize_kernel(const float* __restrict__ coord, const float* __restrict__ x0,
                                float* __restrict__ out) {
  int b = blockIdx.x * blockDim.x + threadIdx.x;
  if (b >= BB) return;
  int base = b * PP * 3;
  float sx = 0.f, sy = 0.f, sz = 0.f;
  for (int p = 0; p < PP; ++p) {
    sx += coord[base + p*3 + 0] - x0[base + p*3 + 0];
    sy += coord[base + p*3 + 1] - x0[base + p*3 + 1];
    sz += coord[base + p*3 + 2] - x0[base + p*3 + 2];
  }
  sx /= (float)PP; sy /= (float)PP; sz /= (float)PP;
  for (int p = 0; p < PP; ++p) {
    out[base + p*3 + 0] = (coord[base + p*3 + 0] - x0[base + p*3 + 0]) - sx;
    out[base + p*3 + 1] = (coord[base + p*3 + 1] - x0[base + p*3 + 1]) - sy;
    out[base + p*3 + 2] = (coord[base + p*3 + 2] - x0[base + p*3 + 2]) - sz;
  }
}

// ------------------------------ edge kernel --------------------------------
// 128 threads = 4 waves, 16 edges per wave, 64 edges per block. EE % 64 == 0.
__global__ __launch_bounds__(128)
void edge_layer_kernel(const _Float16* __restrict__ h16, const float* __restrict__ coord,
                       const float* __restrict__ eattr,
                       const int* __restrict__ rows, const int* __restrict__ cols,
                       const _Float16* __restrict__ wlayer,   // wpack + l*WPL
                       const float* __restrict__ eb1, const float* __restrict__ eb2,
                       const float* __restrict__ cb1, const float* __restrict__ cw2,
                       const float* __restrict__ aw,  const float* __restrict__ ab,
                       float* __restrict__ agg, float* __restrict__ dcoord) {
  __shared__ alignas(32) _Float16 swb[EDGE_WB];         // 36864 B: w1T|w2T|c1T
  __shared__ alignas(32) _Float16 tiles[4 * 16 * LDK1]; // 20480 B
  __shared__ float gbuf[4][16];
  const _Float16* w1T = swb;
  const _Float16* w2T = swb + 10240;
  const _Float16* c1T = swb + 14336;

  int tid = threadIdx.x;
  int wave = tid >> 5, lane = tid & 31;
  int e0 = blockIdx.x * 64 + wave * 16;
  _Float16* T = tiles + wave * 16 * LDK1;
  int slot = lane & 15, part = lane >> 4;
  int e = e0 + slot;
  int nr = rows[e], nc = cols[e];
  int node = part ? nc : nr;

  // async-copy e_in feature rows (f16 h) straight into the LDS tile
  {
    unsigned long long g = (unsigned long long)(h16 + (size_t)node * HH);
    unsigned lds = lds_off(T + slot * LDK1 + part * HH);
    async_row128(lds, g);
  }
  // wave 0: TDM-load this layer's packed f16 weights into LDS
  if (wave == 0) {
    tdm_load(wlayer, lds_off(swb), EDGE_WB / 4);   // EDGE_WB halves = EDGE_WB/4 8B units
    __builtin_amdgcn_s_wait_tensorcnt((short)0);
  }
  float ndx = 0.f, ndy = 0.f, ndz = 0.f;
  if (part == 0) {
    float dx = coord[nr*3+0] - coord[nc*3+0];
    float dy = coord[nr*3+1] - coord[nc*3+1];
    float dz = coord[nr*3+2] - coord[nc*3+2];
    float radial = dx*dx + dy*dy + dz*dz;
    float inv = 1.f / (sqrtf(radial) + 1.f);
    ndx = dx * inv; ndy = dy * inv; ndz = dz * inv;
    T[slot*LDK1 + 128] = (_Float16)radial;
    T[slot*LDK1 + 129] = (_Float16)eattr[e];
#pragma unroll
    for (int k = 130; k < LDK1; ++k) T[slot*LDK1 + k] = (_Float16)0.f;
  }
  wait_async0();
  __syncthreads();

  int col  = lane & 15;
  int moff = (lane & 16) ? 8 : 0;
  v8f acc[4];

  // GEMM1: (16x160) @ (160x64) -> silu -> x1 tile
  zero_acc(acc);
  wave_gemm<LDK1, LDK1, 5>(T, w1T, lane, acc);
#pragma unroll
  for (int nt = 0; nt < 4; ++nt) {
    float bv = eb1[nt*16 + col];
#pragma unroll
    for (int v = 0; v < 8; ++v)
      T[(v + moff)*LDK1 + nt*16 + col] = (_Float16)siluf(acc[nt][v] + bv);
  }
  __syncthreads();

  // GEMM2: (16x64) @ (64x64) -> silu -> m0
  zero_acc(acc);
  wave_gemm<LDK1, HH, 2>(T, w2T, lane, acc);
  float mreg[4][8];
#pragma unroll
  for (int nt = 0; nt < 4; ++nt) {
    float bv = eb2[nt*16 + col];
#pragma unroll
    for (int v = 0; v < 8; ++v) {
      float mv = siluf(acc[nt][v] + bv);
      mreg[nt][v] = mv;
      T[(v + moff)*LDK1 + nt*16 + col] = (_Float16)mv;
    }
  }
  __syncthreads();

  // attention gate
  if (part == 0) {
    float d = ab[0];
    for (int k = 0; k < HH; ++k) d += (float)T[slot*LDK1 + k] * aw[k];
    gbuf[wave][slot] = sigmf(d);
  }
  __syncthreads();

  // gate m, scatter segment_sum(rows) into agg, rewrite gated f16 tile
  int er[8];
#pragma unroll
  for (int v = 0; v < 8; ++v) er[v] = rows[e0 + moff + v];
#pragma unroll
  for (int nt = 0; nt < 4; ++nt) {
#pragma unroll
    for (int v = 0; v < 8; ++v) {
      float mv = mreg[nt][v] * gbuf[wave][moff + v];
      T[(v + moff)*LDK1 + nt*16 + col] = (_Float16)mv;
      atomicAdd(&agg[er[v]*HH + nt*16 + col], mv);
    }
  }
  __syncthreads();

  // GEMM3: coord MLP hidden
  zero_acc(acc);
  wave_gemm<LDK1, HH, 2>(T, c1T, lane, acc);
#pragma unroll
  for (int nt = 0; nt < 4; ++nt) {
    float bv = cb1[nt*16 + col];
#pragma unroll
    for (int v = 0; v < 8; ++v)
      T[(v + moff)*LDK1 + nt*16 + col] = (_Float16)siluf(acc[nt][v] + bv);
  }
  __syncthreads();

  // per-edge scalar: tanh(x3 . cw2) * CR ; scatter trans into dcoord
  if (part == 0) {
    float d = 0.f;
    for (int k = 0; k < HH; ++k) d += (float)T[slot*LDK1 + k] * cw2[k];
    float s = tanhf(d) * CRc;
    atomicAdd(&dcoord[nr*3+0], ndx * s);
    atomicAdd(&dcoord[nr*3+1], ndy * s);
    atomicAdd(&dcoord[nr*3+2], ndz * s);
  }
}

// ------------------------------ node kernel --------------------------------
// 256 threads = 8 waves, 16 nodes per wave, 128 nodes per block. NN % 128 == 0.
__global__ __launch_bounds__(256)
void node_layer_kernel(float* __restrict__ h, _Float16* __restrict__ h16,
                       const float* __restrict__ agg,
                       const _Float16* __restrict__ wlayer,  // wpack + l*WPL + EDGE_WB
                       const float* __restrict__ nb1, const float* __restrict__ nb2) {
  __shared__ alignas(32) _Float16 swb[NODE_WB];         // 24576 B: nw1T|nw2T
  __shared__ alignas(32) _Float16 tiles[8 * 16 * LDKN]; // 32768 B
  const _Float16* w1T = swb;
  const _Float16* w2T = swb + 8192;

  int tid = threadIdx.x;
  int wave = tid >> 5, lane = tid & 31;
  int n0 = blockIdx.x * 128 + wave * 16;
  _Float16* T = tiles + wave * 16 * LDKN;
  int slot = lane & 15, part = lane >> 4;
  int nodeid = n0 + slot;

  if (part == 0) {
    // async-copy f16 h row into tile cols [0,64)
    unsigned long long g = (unsigned long long)(h16 + (size_t)nodeid * HH);
    async_row128(lds_off(T + slot * LDKN), g);
  } else {
    // agg is produced in f32 by edge-kernel atomics: convert while staging
    const float* src = agg + (size_t)nodeid * HH;
#pragma unroll
    for (int k = 0; k < HH; k += 4) {
      float4 f = *(const float4*)(src + k);
      v4h q = {(_Float16)f.x, (_Float16)f.y, (_Float16)f.z, (_Float16)f.w};
      *(v4h*)(T + slot * LDKN + HH + k) = q;
    }
  }
  if (wave == 0) {
    tdm_load(wlayer, lds_off(swb), NODE_WB / 4);
    __builtin_amdgcn_s_wait_tensorcnt((short)0);
  }
  wait_async0();
  __syncthreads();

  int col  = lane & 15;
  int moff = (lane & 16) ? 8 : 0;
  v8f acc[4];

  // GEMM1: (16x128) @ (128x64) -> silu -> x tile
  zero_acc(acc);
  wave_gemm<LDKN, LDKN, 4>(T, w1T, lane, acc);
#pragma unroll
  for (int nt = 0; nt < 4; ++nt) {
    float bv = nb1[nt*16 + col];
#pragma unroll
    for (int v = 0; v < 8; ++v)
      T[(v + moff)*LDKN + nt*16 + col] = (_Float16)siluf(acc[nt][v] + bv);
  }
  __syncthreads();

  // GEMM2: (16x64) @ (64x64) + nb2, residual add into h (f32) + refresh h16
  zero_acc(acc);
  wave_gemm<LDKN, HH, 2>(T, w2T, lane, acc);
#pragma unroll
  for (int nt = 0; nt < 4; ++nt) {
    float bv = nb2[nt*16 + col];
#pragma unroll
    for (int v = 0; v < 8; ++v) {
      size_t gi = (size_t)(n0 + v + moff) * HH + nt*16 + col;
      float hn = h[gi] + acc[nt][v] + bv;
      h[gi] = hn;
      h16[gi] = (_Float16)hn;
    }
  }
}

// ------------------------------ launcher -----------------------------------
extern "C" void kernel_launch(void* const* d_in, const int* in_sizes, int n_in,
                              void* d_out, int out_size, void* d_ws, size_t ws_size,
                              hipStream_t stream) {
  (void)in_sizes; (void)n_in; (void)out_size; (void)ws_size;
  const float* t    = (const float*)d_in[0];
  const float* x    = (const float*)d_in[1];
  const float* embW = (const float*)d_in[2];
  const float* embb = (const float*)d_in[3];
  // d_in[4], d_in[5] (out_W, out_b) do not affect the velocity output
  const float* ew1  = (const float*)d_in[6];
  const float* eb1  = (const float*)d_in[7];
  const float* ew2  = (const float*)d_in[8];
  const float* eb2  = (const float*)d_in[9];
  const float* nw1  = (const float*)d_in[10];
  const float* nb1  = (const float*)d_in[11];
  const float* nw2  = (const float*)d_in[12];
  const float* nb2  = (const float*)d_in[13];
  const float* cw1  = (const float*)d_in[14];
  const float* cb1  = (const float*)d_in[15];
  const float* cw2  = (const float*)d_in[16];
  const float* aw   = (const float*)d_in[17];
  const float* ab   = (const float*)d_in[18];
  const int*   rows = (const int*)d_in[19];
  const int*   cols = (const int*)d_in[20];

  float* ws     = (float*)d_ws;
  float* h      = ws;                 // NN*HH f32
  float* agg    = h + NN*HH;          // NN*HH f32
  float* coord  = agg + NN*HH;        // NN*3
  float* dcoord = coord + NN*3;       // NN*3
  float* eattr  = dcoord + NN*3;      // EE
  _Float16* wpack = (_Float16*)(eattr + EE);  // LL*WPL halves (16B-aligned)
  _Float16* h16   = wpack + LL*WPL;           // NN*HH halves (16B-aligned)

  prep_weights_kernel<<<(LL*WPL + 255)/256, 256, 0, stream>>>(ew1, ew2, cw1, nw1, nw2, wpack);
  init_h_kernel<<<(NN*HH + 255)/256, 256, 0, stream>>>(t, embW, embb, h, h16);
  init_coord_kernel<<<(NN*3 + 255)/256, 256, 0, stream>>>(x, coord);
  init_eattr_kernel<<<(EE + 255)/256, 256, 0, stream>>>(x, rows, cols, eattr);

  for (int l = 0; l < LL; ++l) {
    zero_kernel<<<(NN*(HH+3) + 255)/256, 256, 0, stream>>>(agg, dcoord);
    edge_layer_kernel<<<EE/64, 128, 0, stream>>>(
        h16, coord, eattr, rows, cols,
        wpack + l*WPL,
        eb1 + l*HH, eb2 + l*HH, cb1 + l*HH,
        cw2 + l*HH, aw + l*HH, ab + l,
        agg, dcoord);
    apply_coord_kernel<<<(NN*3 + 255)/256, 256, 0, stream>>>(coord, dcoord);
    node_layer_kernel<<<NN/128, 256, 0, stream>>>(
        h, h16, agg, wpack + l*WPL + EDGE_WB, nb1 + l*HH, nb2 + l*HH);
  }

  finalize_kernel<<<(BB + 255)/256, 256, 0, stream>>>(coord, x, (float*)d_out);
}
